// TransformerLayer_63161789055050
// MI455X (gfx1250) — matrix operand order repaired and checked by reading, not compiled, not run on hardware
//
#include <hip/hip_runtime.h>
#include <hip/hip_bf16.h>

typedef __attribute__((ext_vector_type(16))) _Float16 v16h;
typedef __attribute__((ext_vector_type(8)))  _Float16 v8h;
typedef __attribute__((ext_vector_type(8)))  float    v8f;
typedef __attribute__((ext_vector_type(4))) unsigned int v4u;
typedef __attribute__((ext_vector_type(4))) int v4i_t;
typedef __attribute__((ext_vector_type(8))) int v8i_t;

#define NHEADS 12
#define SEQ    1024
#define HDIM   768
#define FFDIM  3072
#define BATCH  8
#define MTOK   (BATCH*SEQ)

#if defined(__gfx1250__) && __has_builtin(__builtin_amdgcn_tensor_load_to_lds) && __has_builtin(__builtin_amdgcn_s_wait_tensorcnt)
#define USE_TDM 1
#else
#define USE_TDM 0
#endif

static __device__ __forceinline__ v8f zero8() {
    v8f z;
#pragma unroll
    for (int i = 0; i < 8; ++i) z[i] = 0.0f;
    return z;
}

// Fast reciprocal: single v_rcp_f32 instead of the IEEE div-scale/fixup chain.
static __device__ __forceinline__ float rcp_fast(float x) {
#if __has_builtin(__builtin_amdgcn_rcpf)
    return __builtin_amdgcn_rcpf(x);
#else
    return 1.0f / x;
#endif
}

// A-fragment (16x32 f16) lane data: {p[0..7], p[16..23]} per CDNA5 16-bit A layout.
static __device__ __forceinline__ v16h load_split8(const _Float16* p) {
    v16h r;
#pragma unroll
    for (int i = 0; i < 8; ++i) { r[i] = p[i]; r[8 + i] = p[16 + i]; }
    return r;
}

// Branch-free erf (Abramowitz-Stegun 7.1.26, |err| < 1.6e-7): no exec-mask
// branches, v_rcp_f32 instead of full-precision division.
static __device__ __forceinline__ float erf_fast(float x) {
    float ax = fabsf(x);
    float t  = rcp_fast(fmaf(0.3275911f, ax, 1.0f));
    float p  = t * (0.254829592f +
               t * (-0.284496736f +
               t * (1.421413741f +
               t * (-1.453152027f +
               t * 1.061405429f))));
    float r = 1.0f - p * __expf(-ax * ax);
    return copysignf(r, x);
}

static __device__ __forceinline__ float gelu_fast(float v) {
    return 0.5f * v * (1.0f + erf_fast(v * 0.70710678118654752f));
}

#if USE_TDM
// Flat LDS pointers: low 32 bits are the LDS byte offset (ISA 10.2 aperture mapping).
static __device__ __forceinline__ unsigned int lds_off_of(const void* p) {
    return (unsigned int)(unsigned long long)p;
}

// TDM: async DMA of a tile_y x tile_x f16 tile from a row-major [dim1][>=dim0]
// tensor (row stride stride_elems, elements) into LDS, packed row-major
// [tile_y][tile_x]. D# per CDNA5 ISA ch.8 (group0 + group1, 2D).
static __device__ __forceinline__ void tdm_load_tile_f16(
    unsigned int lds_off, const _Float16* gsrc,
    int tile_x, int tile_y, int stride_elems, int dim0, int dim1) {
    unsigned long long ga = (unsigned long long)gsrc;
    v4u g0;
    g0[0] = 1u;                                            // count=1, user mode
    g0[1] = lds_off;                                       // lds_addr (bytes)
    g0[2] = (unsigned int)(ga & 0xffffffffull);            // global_addr[31:0]
    g0[3] = (unsigned int)((ga >> 32) & 0x01ffffffull)     // global_addr[56:32]
            | (2u << 30);                                  // type=2 ("image")
    v8i_t g1;
    g1[0] = (int)(1u << 16);                               // wg_mask=0, data_size=1 (2B)
    g1[1] = (int)(((unsigned)dim0 & 0xffffu) << 16);       // tensor_dim0[15:0]
    g1[2] = (int)((((unsigned)dim0 >> 16) & 0xffffu)       // tensor_dim0[31:16]
            | (((unsigned)dim1 & 0xffffu) << 16));         // tensor_dim1[15:0]
    g1[3] = (int)((((unsigned)dim1 >> 16) & 0xffffu)       // tensor_dim1[31:16]
            | (((unsigned)tile_x & 0xffffu) << 16));       // tile_dim0
    g1[4] = (int)((unsigned)tile_y & 0xffffu);             // tile_dim1 (tile_dim2=0)
    g1[5] = stride_elems;                                  // tensor_dim0_stride[31:0]
    g1[6] = 0;
    g1[7] = 0;
    v4i_t z4; z4[0] = 0; z4[1] = 0; z4[2] = 0; z4[3] = 0;
#if __clang_major__ >= 23
    v8i_t z8;
#pragma unroll
    for (int i = 0; i < 8; ++i) z8[i] = 0;
    __builtin_amdgcn_tensor_load_to_lds(g0, g1, z4, z4, z8, 0);
#else
    __builtin_amdgcn_tensor_load_to_lds(g0, g1, z4, z4, 0);
#endif
}
#endif // USE_TDM

// ---------------------------------------------------------------- converts
__global__ void cvt_f16_kernel(const float* __restrict__ in, _Float16* __restrict__ out, int n) {
    int i = blockIdx.x * blockDim.x + threadIdx.x;
    if (i < n) out[i] = (_Float16)in[i];
}

// W[K][N] f32  ->  Wt[N][K] f16
__global__ void transpose_cvt_kernel(const float* __restrict__ W, _Float16* __restrict__ Wt,
                                     int K, int N) {
    int i = blockIdx.x * blockDim.x + threadIdx.x;
    if (i < K * N) {
        int k = i / N, n = i - k * N;
        Wt[(size_t)n * K + k] = (_Float16)W[i];
    }
}

// ---------------------------------------------------------------- WMMA GEMM (TDM-fed)
// 8 waves/WG; workgroup tile M=256 x N=64; each wave owns 32 rows (8 accums).
// B tile (64x32 f16) streamed global->LDS by the TDM, double-buffered.
// A fragments global->VGPR, software-pipelined one k-chunk ahead.
__global__ __launch_bounds__(256)
void gemm_wmma_kernel(const _Float16* __restrict__ A, const _Float16* __restrict__ Bt,
                      const float* __restrict__ bias, int N, int K,
                      int act, int map,
                      _Float16* __restrict__ out16, float* __restrict__ out32) {
    __shared__ _Float16 bbuf[2][64 * 32];

    const int tid  = threadIdx.x;
    const int wave = tid >> 5;
    const int lane = tid & 31;
    const int sub  = lane >> 4;
    const int col  = lane & 15;
    const int rowbase = blockIdx.x * 256 + wave * 32;
    const int nbase   = blockIdx.y * 64;
    const int nc = K / 32;

    v8f acc[2][4];
#pragma unroll
    for (int m = 0; m < 2; ++m)
#pragma unroll
        for (int t = 0; t < 4; ++t) acc[m][t] = zero8();

    const _Float16* arow0 = A + (size_t)(rowbase + col) * K;
    const _Float16* arow1 = A + (size_t)(rowbase + 16 + col) * K;

#if USE_TDM
    if (wave == 0) {
        tdm_load_tile_f16(lds_off_of(&bbuf[0][0]), Bt + (size_t)nbase * K,
                          32, 64, K, K, N - nbase);
        tdm_load_tile_f16(lds_off_of(&bbuf[1][0]), Bt + (size_t)nbase * K + 32,
                          32, 64, K, K, N - nbase);
    }
#endif

    v16h af0 = load_split8(arow0 + sub * 8);
    v16h af1 = load_split8(arow1 + sub * 8);

    for (int i = 0; i < nc; ++i) {
        const int kc = i * 32;
#if USE_TDM
        const _Float16* bb = &bbuf[i & 1][0];
        if (wave == 0) {
            if (i < nc - 1) __builtin_amdgcn_s_wait_tensorcnt(1);
            else            __builtin_amdgcn_s_wait_tensorcnt(0);
        }
        __syncthreads();
#else
        const _Float16* bb = &bbuf[0][0];
        __syncthreads();
        {   // cooperative load: 64 rows x 32 halves, 16B per thread
            int r  = tid >> 2;
            int ch = (tid & 3) * 8;
            *(v8h*)(&bbuf[0][r * 32 + ch]) =
                *(const v8h*)(Bt + (size_t)(nbase + r) * K + kc + ch);
        }
        __syncthreads();
#endif
        // prefetch next chunk's A fragments (overlaps WMMAs)
        const int kn = (i + 1 < nc) ? kc + 32 : kc;
        v16h an0 = load_split8(arow0 + kn + sub * 8);
        v16h an1 = load_split8(arow1 + kn + sub * 8);

        // batch all B-fragments, then run 8 independent WMMAs back-to-back
        v16h bf[4];
#pragma unroll
        for (int t = 0; t < 4; ++t)
            bf[t] = load_split8(bb + (t * 16 + col) * 32 + sub * 8);
#pragma unroll
        for (int t = 0; t < 4; ++t) {
            acc[0][t] = __builtin_amdgcn_wmma_f32_16x16x32_f16(
                false, af0, false, bf[t], (short)0, acc[0][t], false, false);
            acc[1][t] = __builtin_amdgcn_wmma_f32_16x16x32_f16(
                false, af1, false, bf[t], (short)0, acc[1][t], false, false);
        }
        af0 = an0; af1 = an1;

#if USE_TDM
        __syncthreads();
        if (wave == 0 && i + 2 < nc)
            tdm_load_tile_f16(lds_off_of(&bbuf[i & 1][0]),
                              Bt + (size_t)nbase * K + kc + 64,
                              32, 64, K, K, N - nbase);
#endif
    }

#pragma unroll
    for (int m = 0; m < 2; ++m) {
#pragma unroll
        for (int t = 0; t < 4; ++t) {
#pragma unroll
            for (int r = 0; r < 8; ++r) {
                int row  = rowbase + m * 16 + sub * 8 + r;
                int nidx = nbase + t * 16 + col;
                float v = acc[m][t][r] + bias[nidx];
                if (act) v = gelu_fast(v);
                if (out16) {
                    if (map == 0) {
                        out16[(size_t)row * N + nidx] = (_Float16)v;
                    } else {
                        int bb2 = row >> 10, s = row & 1023;   // S = 1024
                        int hh = nidx >> 6, d = nidx & 63;     // DH = 64
                        if (map == 1)
                            out16[((size_t)(bb2 * NHEADS + hh) * SEQ + s) * 64 + d] = (_Float16)v;
                        else
                            out16[((size_t)(bb2 * NHEADS + hh) * 64 + d) * SEQ + s] = (_Float16)v;
                    }
                }
                if (out32) out32[(size_t)row * N + nidx] = v;
            }
        }
    }
}

// ---------------------------------------------------------------- flash attention
// One wave per (b, h, 16 q-rows). K/V tiles streamed by the TDM, double-buffered,
// pipelined two key-blocks deep. Online softmax; scores never touch HBM.
__global__ __launch_bounds__(32)
void attn_kernel(const _Float16* __restrict__ q16, const _Float16* __restrict__ k16,
                 const _Float16* __restrict__ vt16, const float* __restrict__ pbias,
                 const float* __restrict__ mask, _Float16* __restrict__ ctx16) {
    const int lane = threadIdx.x;
    const int sub  = lane >> 4;
    const int col  = lane & 15;
    const int qt = blockIdx.x, h = blockIdx.y, b = blockIdx.z;
    const int qbase = qt * 16;

    const _Float16* Q  = q16  + (size_t)(b * NHEADS + h) * SEQ * 64;
    const _Float16* Kk = k16  + (size_t)(b * NHEADS + h) * SEQ * 64;
    const _Float16* Vt = vt16 + (size_t)(b * NHEADS + h) * 64 * SEQ;
    const float*    pb = pbias + ((size_t)h * SEQ + qbase) * SEQ;
    const float*    mk = mask  + ((size_t)b * SEQ + qbase) * SEQ;

#if USE_TDM
    __shared__ _Float16 kbuf[2][32 * 64];   // [key][d]
    __shared__ _Float16 vbuf[2][64 * 32];   // [d][key]
#endif
    __shared__ _Float16 lds_p[16 * 32];

    v16h qfrag[2];
#pragma unroll
    for (int c = 0; c < 2; ++c)
        qfrag[c] = load_split8(Q + (size_t)(qbase + col) * 64 + c * 32 + sub * 8);

    v8f cacc[4];
#pragma unroll
    for (int t = 0; t < 4; ++t) cacc[t] = zero8();

    float m_run[8], l_run[8];
#pragma unroll
    for (int r = 0; r < 8; ++r) { m_run[r] = -1e30f; l_run[r] = 0.0f; }

#if USE_TDM
    tdm_load_tile_f16(lds_off_of(&kbuf[0][0]), Kk,           64, 32, 64,  64, SEQ);
    tdm_load_tile_f16(lds_off_of(&vbuf[0][0]), Vt,           32, 64, SEQ, SEQ, 64);
    tdm_load_tile_f16(lds_off_of(&kbuf[1][0]), Kk + 32 * 64, 64, 32, 64,  64, SEQ);
    tdm_load_tile_f16(lds_off_of(&vbuf[1][0]), Vt + 32,      32, 64, SEQ, SEQ, 64);
#endif

    const int nkb = SEQ / 32;
    for (int i = 0; i < nkb; ++i) {
        const int kb = i * 32;

        // one instruction prefetches all of next block's bias/mask cache lines:
        // lanes 0-15 cover position_bias rows, lanes 16-31 cover mask rows
        if (kb + 32 < SEQ) {
            const float* pf = (lane < 16) ? (pb + (size_t)lane * SEQ + kb + 32)
                                          : (mk + (size_t)(lane - 16) * SEQ + kb + 32);
            __builtin_prefetch(pf, 0, 1);
        }

#if USE_TDM
        if (i < nkb - 1) __builtin_amdgcn_s_wait_tensorcnt(2);
        else             __builtin_amdgcn_s_wait_tensorcnt(0);
        const _Float16* kbp = &kbuf[i & 1][0];
        const _Float16* vbp = &vbuf[i & 1][0];
#endif

        // ---- scores: Q(16x64) @ K^T(64x32) -> two 16x16 f32 tiles
        v16h kf[2][2];
#pragma unroll
        for (int t = 0; t < 2; ++t)
#pragma unroll
            for (int c = 0; c < 2; ++c)
#if USE_TDM
                kf[t][c] = load_split8(kbp + (t * 16 + col) * 64 + c * 32 + sub * 8);
#else
                kf[t][c] = load_split8(Kk + (size_t)(kb + t * 16 + col) * 64 + c * 32 + sub * 8);
#endif
        v8f sacc[2];
        sacc[0] = zero8(); sacc[1] = zero8();
#pragma unroll
        for (int t = 0; t < 2; ++t)
#pragma unroll
            for (int c = 0; c < 2; ++c)
                sacc[t] = __builtin_amdgcn_wmma_f32_16x16x32_f16(
                    false, qfrag[c], false, kf[t][c], (short)0, sacc[t], false, false);

        // ---- bias/scale/mask + online softmax
        float p0[8], p1[8];
#pragma unroll
        for (int r = 0; r < 8; ++r) {
            int rrow = sub * 8 + r;
            int key0 = kb + col, key1 = kb + 16 + col;
            float s0 = (sacc[0][r] + pb[(size_t)rrow * SEQ + key0]) * 0.125f
                       + mk[(size_t)rrow * SEQ + key0];
            float s1 = (sacc[1][r] + pb[(size_t)rrow * SEQ + key1]) * 0.125f
                       + mk[(size_t)rrow * SEQ + key1];
            float v = fmaxf(s0, s1);
#pragma unroll
            for (int mm = 1; mm < 16; mm <<= 1) v = fmaxf(v, __shfl_xor(v, mm, 32));
            float mn = fmaxf(m_run[r], v);
            float sc = __expf(m_run[r] - mn);
            m_run[r] = mn;
            float e0 = __expf(s0 - mn), e1 = __expf(s1 - mn);
            float rs = e0 + e1;
#pragma unroll
            for (int mm = 1; mm < 16; mm <<= 1) rs += __shfl_xor(rs, mm, 32);
            l_run[r] = l_run[r] * sc + rs;
            p0[r] = e0; p1[r] = e1;
#pragma unroll
            for (int t = 0; t < 4; ++t) cacc[t][r] *= sc;
        }

        // ---- re-shuffle probs (C layout -> A layout) through LDS
        __syncthreads();
#pragma unroll
        for (int r = 0; r < 8; ++r) {
            int rrow = sub * 8 + r;
            lds_p[rrow * 32 + col]      = (_Float16)p0[r];
            lds_p[rrow * 32 + 16 + col] = (_Float16)p1[r];
        }
        __syncthreads();
        v16h afrag = load_split8(lds_p + col * 32 + sub * 8);

        // ---- ctx += P(16x32) @ V(32x64)
        v16h vf[4];
#pragma unroll
        for (int t = 0; t < 4; ++t)
#if USE_TDM
            vf[t] = load_split8(vbp + (t * 16 + col) * 32 + sub * 8);
#else
            vf[t] = load_split8(Vt + (size_t)(t * 16 + col) * SEQ + kb + sub * 8);
#endif
#pragma unroll
        for (int t = 0; t < 4; ++t)
            cacc[t] = __builtin_amdgcn_wmma_f32_16x16x32_f16(
                false, afrag, false, vf[t], (short)0, cacc[t], false, false);

#if USE_TDM
        __syncthreads();   // ds reads retired before TDM may overwrite buffers
        if (i + 2 < nkb) {
            tdm_load_tile_f16(lds_off_of(&kbuf[i & 1][0]),
                              Kk + (size_t)(kb + 64) * 64, 64, 32, 64, 64, SEQ);
            tdm_load_tile_f16(lds_off_of(&vbuf[i & 1][0]),
                              Vt + kb + 64, 32, 64, SEQ, SEQ, 64);
        }
#endif
    }

#pragma unroll
    for (int r = 0; r < 8; ++r) {
        int rrow = sub * 8 + r;
        float inv = rcp_fast(l_run[r]);
        size_t base = ((size_t)(b * SEQ + qbase + rrow)) * HDIM + h * 64;
#pragma unroll
        for (int t = 0; t < 4; ++t)
            ctx16[base + t * 16 + col] = (_Float16)(cacc[t][r] * inv);
    }
}

// ---------------------------------------------------------------- residual + LayerNorm
__global__ __launch_bounds__(256)
void add_ln_kernel(const float* __restrict__ x, const float* __restrict__ res,
                   const float* __restrict__ g, const float* __restrict__ bta,
                   float* __restrict__ o32, _Float16* __restrict__ o16, int N) {
    int row = blockIdx.x, tid = threadIdx.x;
    const float* xr = x   + (size_t)row * N;
    const float* rr = res + (size_t)row * N;

    float y[3];
    float s = 0.0f, s2 = 0.0f;
#pragma unroll
    for (int i = 0; i < 3; ++i) {
        int c = tid + i * 256;
        y[i] = xr[c] + rr[c];
        s += y[i]; s2 += y[i] * y[i];
    }
#pragma unroll
    for (int mm = 16; mm >= 1; mm >>= 1) {
        s  += __shfl_xor(s,  mm, 32);
        s2 += __shfl_xor(s2, mm, 32);
    }
    __shared__ float sh[16];
    __shared__ float stats[2];
    int wid = tid >> 5, ln = tid & 31;
    if (ln == 0) { sh[wid] = s; sh[8 + wid] = s2; }
    __syncthreads();
    if (tid == 0) {
        float a = 0.0f, b2 = 0.0f;
        for (int w = 0; w < 8; ++w) { a += sh[w]; b2 += sh[8 + w]; }
        stats[0] = a; stats[1] = b2;
    }
    __syncthreads();
    float mean = stats[0] / (float)N;
    float var  = (stats[1] - stats[0] * stats[0] / (float)N) / (float)(N - 1);
    float inv  = 1.0f / (sqrtf(fmaxf(var, 0.0f)) + 1e-6f);
#pragma unroll
    for (int i = 0; i < 3; ++i) {
        int c = tid + i * 256;
        float o = g[c] * (y[i] - mean) * inv + bta[c];
        o32[(size_t)row * N + c] = o;
        if (o16) o16[(size_t)row * N + c] = (_Float16)o;
    }
}

// ---------------------------------------------------------------- host
extern "C" void kernel_launch(void* const* d_in, const int* in_sizes, int n_in,
                              void* d_out, int out_size, void* d_ws, size_t ws_size,
                              hipStream_t stream) {
    (void)in_sizes; (void)n_in; (void)out_size; (void)ws_size;

    const float* hidden = (const float*)d_in[0];
    const float* mask   = (const float*)d_in[1];
    const float* pbias  = (const float*)d_in[2];
    const float* Wq = (const float*)d_in[3];  const float* bq = (const float*)d_in[4];
    const float* Wk = (const float*)d_in[5];  const float* bk = (const float*)d_in[6];
    const float* Wv = (const float*)d_in[7];  const float* bv = (const float*)d_in[8];
    const float* Wo = (const float*)d_in[9];  const float* bo = (const float*)d_in[10];
    const float* W1 = (const float*)d_in[11]; const float* b1 = (const float*)d_in[12];
    const float* W2 = (const float*)d_in[13]; const float* b2 = (const float*)d_in[14];
    const float* g1v = (const float*)d_in[15]; const float* be1 = (const float*)d_in[16];
    const float* g2v = (const float*)d_in[17]; const float* be2 = (const float*)d_in[18];

    uint8_t* w = (uint8_t*)d_ws;
    auto take = [&](size_t bytes) {
        uint8_t* p = w;
        w += (bytes + 255) & ~(size_t)255;
        return p;
    };
    _Float16* h16     = (_Float16*)take((size_t)MTOK * HDIM * 2);
    _Float16* wqt     = (_Float16*)take((size_t)HDIM * HDIM * 2);
    _Float16* wkt     = (_Float16*)take((size_t)HDIM * HDIM * 2);
    _Float16* wvt     = (_Float16*)take((size_t)HDIM * HDIM * 2);
    _Float16* wot     = (_Float16*)take((size_t)HDIM * HDIM * 2);
    _Float16* w1t     = (_Float16*)take((size_t)FFDIM * HDIM * 2);
    _Float16* w2t     = (_Float16*)take((size_t)HDIM * FFDIM * 2);
    _Float16* q16     = (_Float16*)take((size_t)MTOK * HDIM * 2);
    _Float16* k16     = (_Float16*)take((size_t)MTOK * HDIM * 2);
    _Float16* vt16    = (_Float16*)take((size_t)MTOK * HDIM * 2);
    _Float16* ctx16   = (_Float16*)take((size_t)MTOK * HDIM * 2);
    float*    attn32  = (float*)take((size_t)MTOK * HDIM * 4);
    float*    inter32 = (float*)take((size_t)MTOK * HDIM * 4);
    _Float16* inter16 = (_Float16*)take((size_t)MTOK * HDIM * 2);
    _Float16* f1_16   = (_Float16*)take((size_t)MTOK * FFDIM * 2);
    float*    f2_32   = (float*)take((size_t)MTOK * HDIM * 4);

    cvt_f16_kernel<<<(MTOK * HDIM + 255) / 256, 256, 0, stream>>>(hidden, h16, MTOK * HDIM);
    transpose_cvt_kernel<<<(HDIM * HDIM + 255) / 256, 256, 0, stream>>>(Wq, wqt, HDIM, HDIM);
    transpose_cvt_kernel<<<(HDIM * HDIM + 255) / 256, 256, 0, stream>>>(Wk, wkt, HDIM, HDIM);
    transpose_cvt_kernel<<<(HDIM * HDIM + 255) / 256, 256, 0, stream>>>(Wv, wvt, HDIM, HDIM);
    transpose_cvt_kernel<<<(HDIM * HDIM + 255) / 256, 256, 0, stream>>>(Wo, wot, HDIM, HDIM);
    transpose_cvt_kernel<<<(HDIM * FFDIM + 255) / 256, 256, 0, stream>>>(W1, w1t, HDIM, FFDIM);
    transpose_cvt_kernel<<<(FFDIM * HDIM + 255) / 256, 256, 0, stream>>>(W2, w2t, FFDIM, HDIM);

    dim3 gH(MTOK / 256, HDIM / 64);    // 32 x 12
    dim3 gF(MTOK / 256, FFDIM / 64);   // 32 x 48

    gemm_wmma_kernel<<<gH, 256, 0, stream>>>(h16, wqt, bq, HDIM, HDIM, 0, 1, q16, nullptr);
    gemm_wmma_kernel<<<gH, 256, 0, stream>>>(h16, wkt, bk, HDIM, HDIM, 0, 1, k16, nullptr);
    gemm_wmma_kernel<<<gH, 256, 0, stream>>>(h16, wvt, bv, HDIM, HDIM, 0, 2, vt16, nullptr);

    attn_kernel<<<dim3(SEQ / 16, NHEADS, BATCH), 32, 0, stream>>>(
        q16, k16, vt16, pbias, mask, ctx16);

    gemm_wmma_kernel<<<gH, 256, 0, stream>>>(ctx16, wot, bo, HDIM, HDIM, 0, 0, nullptr, attn32);
    add_ln_kernel<<<MTOK, 256, 0, stream>>>(attn32, hidden, g1v, be1, inter32, inter16, HDIM);
    gemm_wmma_kernel<<<gF, 256, 0, stream>>>(inter16, w1t, b1, FFDIM, HDIM, 1, 0, f1_16, nullptr);
    gemm_wmma_kernel<<<gH, 256, 0, stream>>>(f1_16, w2t, b2, HDIM, FFDIM, 0, 0, nullptr, f2_32);
    add_ln_kernel<<<MTOK, 256, 0, stream>>>(f2_32, inter32, g2v, be2, (float*)d_out, nullptr, HDIM);
}
